// TransformerBlock_68186900791884
// MI455X (gfx1250) — matrix operand order repaired and checked by reading, not compiled, not run on hardware
//
#include <hip/hip_runtime.h>
#include <math.h>

// ---------------------------------------------------------------------------
// Point-Transformer block for MI455X (gfx1250, wave32, WMMA bf16 16x16x32)
// T=1, B=4, N=4096, K=16, DP=64, DM=128
// ---------------------------------------------------------------------------

#define NB   4        // batches
#define NPTS 4096     // points
#define KNN  16       // neighbors
#define DP   64
#define DM   128

typedef __attribute__((ext_vector_type(16))) __bf16 v16bf;
typedef __attribute__((ext_vector_type(8)))  float  v8f;

static __device__ __forceinline__ v8f wmma_bf16(v16bf a, v16bf b, v8f c) {
  // D = A(16x32 bf16) * B(32x16 bf16) + C(16x16 f32)
  return __builtin_amdgcn_wmma_f32_16x16x32_bf16(
      /*neg_a=*/false, a, /*neg_b=*/false, b,
      /*c_mod=*/(short)0, c, /*reuse_a=*/false, /*reuse_b=*/false);
}

// ---------------------------------------------------------------------------
// Workspace map (all pointers into d_ws)
// ---------------------------------------------------------------------------
struct WS {
  int*    knn;                                   // (B,N,K)
  float*  q;  float* k;  float* v;  float* res;  // (B,DM,N) each
  __bf16* fc1pk; __bf16* wqpk; __bf16* wkpk; __bf16* wvpk;
  __bf16* g1pk;  __bf16* g2pk; __bf16* d2pk; __bf16* fc2pk;
  float*  fc1eb; float* qeb; float* keb; float* veb;
  float*  g1eb;  float* g2eb; float* d2eb; float* d1eb; float* fc2eb;
  float*  d1w;                                   // folded d1 weights (128x3) fp32
};

struct PackArgs {
  const float* w[9]; const float* b[9]; const float* g[9]; const float* be[9];
};

// ---------------------------------------------------------------------------
// Weight packing: fold BN, convert to bf16, pre-swizzle into A-fragment layout.
// A fragment (16x32, MxK): lane l (0..31), half h (0..15):
//   m = 16*mt + (l&15)
//   k = 32*kt + ((l&16)?16:0) + h
// Stored per-fragment, per-lane contiguous: dst[((mt*KT+kt)*32 + l)*16 + h]
// ---------------------------------------------------------------------------
static __device__ void packA(const float* W, const float* g, __bf16* dst,
                             int M, int K, int tid, int nthr) {
  const int KT = K >> 5;
  const int total = M * K;
  for (int e = tid; e < total; e += nthr) {
    int h    = e & 15;
    int lane = (e >> 4) & 31;
    int f    = e >> 9;
    int kt   = f % KT;
    int mt   = f / KT;
    int m = mt * 16 + (lane & 15);
    int k = kt * 32 + ((lane & 16) ? 16 : 0) + h;
    dst[e] = (__bf16)(g[m] * W[m * K + k]);
  }
}

static __device__ void packEB(const float* b, const float* g, const float* be,
                              float* dst, int M, int tid, int nthr) {
  for (int m = tid; m < M; m += nthr) dst[m] = g[m] * b[m] + be[m];
}

__global__ __launch_bounds__(256) void pack_kernel(PackArgs pa, WS ws) {
  const int tid = threadIdx.x, NT = 256;
  // order in pa: 0=fc1 1=wq 2=wk 3=wv 4=g1 5=g2 6=d1 7=d2 8=fc2
  packA(pa.w[0], pa.g[0], ws.fc1pk, DM, DP, tid, NT);
  packA(pa.w[1], pa.g[1], ws.wqpk,  DM, DM, tid, NT);
  packA(pa.w[2], pa.g[2], ws.wkpk,  DM, DM, tid, NT);
  packA(pa.w[3], pa.g[3], ws.wvpk,  DM, DM, tid, NT);
  packA(pa.w[4], pa.g[4], ws.g1pk,  DM, DM, tid, NT);
  packA(pa.w[5], pa.g[5], ws.g2pk,  DM, DM, tid, NT);
  packA(pa.w[7], pa.g[7], ws.d2pk,  DM, DM, tid, NT);
  packA(pa.w[8], pa.g[8], ws.fc2pk, DP, DM, tid, NT);
  packEB(pa.b[0], pa.g[0], pa.be[0], ws.fc1eb, DM, tid, NT);
  packEB(pa.b[1], pa.g[1], pa.be[1], ws.qeb,   DM, tid, NT);
  packEB(pa.b[2], pa.g[2], pa.be[2], ws.keb,   DM, tid, NT);
  packEB(pa.b[3], pa.g[3], pa.be[3], ws.veb,   DM, tid, NT);
  packEB(pa.b[4], pa.g[4], pa.be[4], ws.g1eb,  DM, tid, NT);
  packEB(pa.b[5], pa.g[5], pa.be[5], ws.g2eb,  DM, tid, NT);
  packEB(pa.b[7], pa.g[7], pa.be[7], ws.d2eb,  DM, tid, NT);
  packEB(pa.b[6], pa.g[6], pa.be[6], ws.d1eb,  DM, tid, NT);
  packEB(pa.b[8], pa.g[8], pa.be[8], ws.fc2eb, DP, tid, NT);
  for (int e = tid; e < DM * 3; e += NT) {       // d1 (128x3) kept fp32 (K=3)
    int c = e / 3;
    ws.d1w[e] = pa.g[6][c] * pa.w[6][e];
  }
}

// ---------------------------------------------------------------------------
// KNN: d2 = |p|^2 + |q|^2 - 2 p.q (same formula as reference), stable 16-best.
// One thread per query point; candidate tiles staged in LDS.
// ---------------------------------------------------------------------------
__global__ __launch_bounds__(256) void knn_kernel(const float* __restrict__ xyz,
                                                  int* __restrict__ knn) {
  const int b = blockIdx.x >> 4;                       // 16 blocks / batch
  const int n = ((blockIdx.x & 15) << 8) + threadIdx.x;
  const float* L = xyz + (size_t)b * NPTS * 3;
  const float px = L[n * 3 + 0], py = L[n * 3 + 1], pz = L[n * 3 + 2];
  const float psq = px * px + py * py + pz * pz;

  float dist[KNN];
  int   idx[KNN];
#pragma unroll
  for (int i = 0; i < KNN; ++i) { dist[i] = 3.4e38f; idx[i] = 0; }

  __shared__ float sx[256], sy[256], sz[256], ssq[256];
  for (int c0 = 0; c0 < NPTS; c0 += 256) {
    __syncthreads();
    const int mload = c0 + threadIdx.x;
    const float mx = L[mload * 3 + 0], my = L[mload * 3 + 1], mz = L[mload * 3 + 2];
    sx[threadIdx.x] = mx; sy[threadIdx.x] = my; sz[threadIdx.x] = mz;
    ssq[threadIdx.x] = mx * mx + my * my + mz * mz;
    __syncthreads();
    for (int mm = 0; mm < 256; ++mm) {
      const float d = psq + ssq[mm] - 2.0f * (px * sx[mm] + py * sy[mm] + pz * sz[mm]);
      if (d < dist[KNN - 1]) {                          // strict <: stable ties
        dist[KNN - 1] = d; idx[KNN - 1] = c0 + mm;
#pragma unroll
        for (int t = KNN - 1; t > 0; --t) {
          if (dist[t] < dist[t - 1]) {
            float td = dist[t]; dist[t] = dist[t - 1]; dist[t - 1] = td;
            int   ti = idx[t];  idx[t]  = idx[t - 1];  idx[t - 1]  = ti;
          } else break;
        }
      }
    }
  }
#pragma unroll
  for (int i = 0; i < KNN; ++i)
    knn[((size_t)b * NPTS + n) * KNN + i] = idx[i];
}

// ---------------------------------------------------------------------------
// fc1 + wq/wk/wv. 4 waves/block, one 16-column n-tile per wave.
// ---------------------------------------------------------------------------
__global__ __launch_bounds__(128) void fc1qkv_kernel(const float* __restrict__ feat,
                                                     WS ws) {
  __shared__ float sX[4][DM][16];                      // 32 KB
  const int tid = threadIdx.x, w = tid >> 5, lane = tid & 31;
  const int j = lane & 15, hb = lane >> 4;
  const int b  = blockIdx.x >> 6;                      // 64 blocks / batch
  const int n  = ((blockIdx.x & 63) << 6) + (w << 4) + j;
  const float* F = feat + ((size_t)b * NPTS + n) * DP;

  // B fragments of features^T (K = DP = 64): lane col = n, k = kt*32 + hb*16 + h
  v16bf fB[2];
#pragma unroll
  for (int kt = 0; kt < 2; ++kt)
#pragma unroll
    for (int h = 0; h < 16; ++h)
      fB[kt][h] = (__bf16)F[kt * 32 + hb * 16 + h];

  const v16bf* fc1A = (const v16bf*)ws.fc1pk;
#pragma unroll
  for (int mt = 0; mt < 8; ++mt) {
    v8f acc = {};
#pragma unroll
    for (int kt = 0; kt < 2; ++kt)
      acc = wmma_bf16(fc1A[(mt * 2 + kt) * 32 + lane], fB[kt], acc);
#pragma unroll
    for (int r = 0; r < 8; ++r) {
      const int m = mt * 16 + r + hb * 8;
      sX[w][m][j] = acc[r] + ws.fc1eb[m];
    }
  }
  __syncthreads();

  // rebuild x as B fragments (K = DM = 128)
  v16bf xB[4];
#pragma unroll
  for (int kt = 0; kt < 4; ++kt)
#pragma unroll
    for (int h = 0; h < 16; ++h)
      xB[kt][h] = (__bf16)sX[w][kt * 32 + hb * 16 + h][j];

  const v16bf* WA[3]  = {(const v16bf*)ws.wqpk, (const v16bf*)ws.wkpk,
                         (const v16bf*)ws.wvpk};
  const float* EB[3]  = {ws.qeb, ws.keb, ws.veb};
  float*       OUT[3] = {ws.q, ws.k, ws.v};
#pragma unroll
  for (int o = 0; o < 3; ++o) {
    const v16bf* A = WA[o];
    float* O = OUT[o] + (size_t)b * DM * NPTS;
#pragma unroll
    for (int mt = 0; mt < 8; ++mt) {
      v8f acc = {};
#pragma unroll
      for (int kt = 0; kt < 4; ++kt)
        acc = wmma_bf16(A[(mt * 4 + kt) * 32 + lane], xB[kt], acc);
#pragma unroll
      for (int r = 0; r < 8; ++r) {
        const int m = mt * 16 + r + hb * 8;
        O[(size_t)m * NPTS + n] = acc[r] + EB[o][m];
      }
    }
  }
}

// ---------------------------------------------------------------------------
// Fused attention: one point per wave (4 waves/block).
// pos = d2(relu(d1(diff)));  t = q - k[idx] + pos;  a = g2(relu(g1(t)))
// attn = softmax_K(a/sqrt(DM));  res = sum_K attn*(v[idx]+pos)
// ---------------------------------------------------------------------------
__global__ __launch_bounds__(128) void attn_kernel(const float* __restrict__ xyz,
                                                   WS ws,
                                                   float* __restrict__ attn_out) {
  __shared__ __bf16 sT[4][DM][16];                     // 16 KB (y1 -> t -> u)
  __shared__ float  sVP[4][DM][16];                    // 32 KB (v[idx]+pos)
  __shared__ float  sDf[4][16][3];
  __shared__ int    sIdx[4][16];

  const int tid = threadIdx.x, w = tid >> 5, lane = tid & 31;
  const int j = lane & 15, hb = lane >> 4;
  const int b = blockIdx.x >> 10;                      // 1024 blocks / batch
  const int n = ((blockIdx.x & 1023) << 2) + w;
  const float* L = xyz + (size_t)b * NPTS * 3;

  if (lane < 16) {
    const int gi = ws.knn[((size_t)b * NPTS + n) * KNN + lane];
    sIdx[w][lane]   = gi;
    sDf[w][lane][0] = L[n * 3 + 0] - L[gi * 3 + 0];
    sDf[w][lane][1] = L[n * 3 + 1] - L[gi * 3 + 1];
    sDf[w][lane][2] = L[n * 3 + 2] - L[gi * 3 + 2];
  }

  // y1 = relu(d1(diff))  -> sT   (K=3: scalar VALU, not worth WMMA)
  for (int e = lane; e < DM * 16; e += 32) {
    const int c = e >> 4, jj = e & 15;
    const float y = ws.d1w[c * 3 + 0] * sDf[w][jj][0]
                  + ws.d1w[c * 3 + 1] * sDf[w][jj][1]
                  + ws.d1w[c * 3 + 2] * sDf[w][jj][2] + ws.d1eb[c];
    sT[w][c][jj] = (__bf16)fmaxf(y, 0.0f);
  }

  v16bf yB[4];
#pragma unroll
  for (int kt = 0; kt < 4; ++kt)
#pragma unroll
    for (int h = 0; h < 16; ++h)
      yB[kt][h] = sT[w][kt * 32 + hb * 16 + h][j];

  // pos = d2 @ y1 + bias   (kept in D fragments)
  const v16bf* d2A = (const v16bf*)ws.d2pk;
  v8f pos[8];
#pragma unroll
  for (int mt = 0; mt < 8; ++mt) {
    v8f acc = {};
#pragma unroll
    for (int kt = 0; kt < 4; ++kt)
      acc = wmma_bf16(d2A[(mt * 4 + kt) * 32 + lane], yB[kt], acc);
#pragma unroll
    for (int r = 0; r < 8; ++r) acc[r] += ws.d2eb[mt * 16 + r + hb * 8];
    pos[mt] = acc;
  }

  // gather k/v, build vp = v[idx]+pos and t = q - k[idx] + pos (overwrites sT)
  const int gi = sIdx[w][j];
  const float* qb = ws.q + (size_t)b * DM * NPTS;
  const float* kb = ws.k + (size_t)b * DM * NPTS;
  const float* vb = ws.v + (size_t)b * DM * NPTS;
#pragma unroll
  for (int mt = 0; mt < 8; ++mt)
#pragma unroll
    for (int r = 0; r < 8; ++r) {
      const int m = mt * 16 + r + hb * 8;
      const float p = pos[mt][r];
      sVP[w][m][j] = vb[(size_t)m * NPTS + gi] + p;
      sT[w][m][j]  = (__bf16)(qb[(size_t)m * NPTS + n]
                              - kb[(size_t)m * NPTS + gi] + p);
    }

  // u = relu(g1 @ t + bias) -> sT
  v16bf tB[4];
#pragma unroll
  for (int kt = 0; kt < 4; ++kt)
#pragma unroll
    for (int h = 0; h < 16; ++h)
      tB[kt][h] = sT[w][kt * 32 + hb * 16 + h][j];
  const v16bf* g1A = (const v16bf*)ws.g1pk;
#pragma unroll
  for (int mt = 0; mt < 8; ++mt) {
    v8f acc = {};
#pragma unroll
    for (int kt = 0; kt < 4; ++kt)
      acc = wmma_bf16(g1A[(mt * 4 + kt) * 32 + lane], tB[kt], acc);
#pragma unroll
    for (int r = 0; r < 8; ++r) {
      const int m = mt * 16 + r + hb * 8;
      sT[w][m][j] = (__bf16)fmaxf(acc[r] + ws.g1eb[m], 0.0f);
    }
  }

  // a = g2 @ u + bias; softmax over K (16-lane group); res = sum attn*vp
  v16bf uB[4];
#pragma unroll
  for (int kt = 0; kt < 4; ++kt)
#pragma unroll
    for (int h = 0; h < 16; ++h)
      uB[kt][h] = sT[w][kt * 32 + hb * 16 + h][j];
  const v16bf* g2A = (const v16bf*)ws.g2pk;
  const float scale = 0.08838834764831845f;            // 1/sqrt(128)
#pragma unroll
  for (int mt = 0; mt < 8; ++mt) {
    v8f acc = {};
#pragma unroll
    for (int kt = 0; kt < 4; ++kt)
      acc = wmma_bf16(g2A[(mt * 4 + kt) * 32 + lane], uB[kt], acc);
#pragma unroll
    for (int r = 0; r < 8; ++r) {
      const int m = mt * 16 + r + hb * 8;
      float x = (acc[r] + ws.g2eb[m]) * scale;         // row lives in 16 lanes
      float mx = x;
#pragma unroll
      for (int o = 1; o < 16; o <<= 1) mx = fmaxf(mx, __shfl_xor(mx, o, 32));
      const float e = __expf(x - mx);
      float s = e;
#pragma unroll
      for (int o = 1; o < 16; o <<= 1) s += __shfl_xor(s, o, 32);
      const float at = e / s;
      attn_out[(((size_t)b * DM + m) * NPTS + n) * KNN + j] = at;
      float rv = at * sVP[w][m][j];
#pragma unroll
      for (int o = 1; o < 16; o <<= 1) rv += __shfl_xor(rv, o, 32);
      if (j == 0) ws.res[((size_t)b * DM + m) * NPTS + n] = rv;
    }
  }
}

// ---------------------------------------------------------------------------
// fc2 (128 -> 64) + residual add of original features.
// ---------------------------------------------------------------------------
__global__ __launch_bounds__(128) void fc2_kernel(const float* __restrict__ feat,
                                                  WS ws, float* __restrict__ out) {
  const int tid = threadIdx.x, w = tid >> 5, lane = tid & 31;
  const int j = lane & 15, hb = lane >> 4;
  const int b = blockIdx.x >> 6;
  const int n = ((blockIdx.x & 63) << 6) + (w << 4) + j;

  const float* R = ws.res + (size_t)b * DM * NPTS;
  v16bf rB[4];
#pragma unroll
  for (int kt = 0; kt < 4; ++kt)
#pragma unroll
    for (int h = 0; h < 16; ++h)
      rB[kt][h] = (__bf16)R[(size_t)(kt * 32 + hb * 16 + h) * NPTS + n];

  const v16bf* A = (const v16bf*)ws.fc2pk;
  const float* F = feat + ((size_t)b * NPTS + n) * DP;
  float*       O = out  + ((size_t)b * NPTS + n) * DP;
#pragma unroll
  for (int mt = 0; mt < 4; ++mt) {
    v8f acc = {};
#pragma unroll
    for (int kt = 0; kt < 4; ++kt)
      acc = wmma_bf16(A[(mt * 4 + kt) * 32 + lane], rB[kt], acc);
#pragma unroll
    for (int r = 0; r < 8; ++r) {
      const int m = mt * 16 + r + hb * 8;
      O[m] = acc[r] + ws.fc2eb[m] + F[m];
    }
  }
}

// ---------------------------------------------------------------------------
// Launch
// ---------------------------------------------------------------------------
extern "C" void kernel_launch(void* const* d_in, const int* in_sizes, int n_in,
                              void* d_out, int out_size, void* d_ws, size_t ws_size,
                              hipStream_t stream) {
  (void)in_sizes; (void)n_in; (void)out_size; (void)ws_size;
  const float* xyz  = (const float*)d_in[0];
  const float* feat = (const float*)d_in[1];

  PackArgs pa;
  for (int i = 0; i < 9; ++i) {
    pa.w[i]  = (const float*)d_in[2 + 4 * i + 0];
    pa.b[i]  = (const float*)d_in[2 + 4 * i + 1];
    pa.g[i]  = (const float*)d_in[2 + 4 * i + 2];
    pa.be[i] = (const float*)d_in[2 + 4 * i + 3];
  }

  // ---- workspace layout (float units) ----
  float* wf = (float*)d_ws;
  const size_t QKV = (size_t)NB * DM * NPTS;           // 2 097 152 floats
  WS ws;
  ws.knn = (int*)d_ws;                                 // 262 144 ints
  size_t off = (size_t)NB * NPTS * KNN;                // in floats
  ws.q   = wf + off; off += QKV;
  ws.k   = wf + off; off += QKV;
  ws.v   = wf + off; off += QKV;
  ws.res = wf + off; off += QKV;
  __bf16* pk = (__bf16*)(wf + off);                    // bf16 packed weights
  ws.fc1pk = pk;           pk += DM * DP;              // 8192
  ws.wqpk  = pk;           pk += DM * DM;
  ws.wkpk  = pk;           pk += DM * DM;
  ws.wvpk  = pk;           pk += DM * DM;
  ws.g1pk  = pk;           pk += DM * DM;
  ws.g2pk  = pk;           pk += DM * DM;
  ws.d2pk  = pk;           pk += DM * DM;
  ws.fc2pk = pk;           pk += DP * DM;
  float* fp = (float*)pk;
  ws.fc1eb = fp; fp += DM;  ws.qeb  = fp; fp += DM;
  ws.keb   = fp; fp += DM;  ws.veb  = fp; fp += DM;
  ws.g1eb  = fp; fp += DM;  ws.g2eb = fp; fp += DM;
  ws.d2eb  = fp; fp += DM;  ws.d1eb = fp; fp += DM;
  ws.fc2eb = fp; fp += DP;  ws.d1w  = fp; fp += DM * 3;

  float* out  = (float*)d_out;                         // (1,B,N,DP)
  float* attn = out + (size_t)NB * NPTS * DP;          // (B,DM,N,K)

  pack_kernel  <<<1,    256, 0, stream>>>(pa, ws);
  knn_kernel   <<<NB * (NPTS / 256), 256, 0, stream>>>(xyz, ws.knn);
  fc1qkv_kernel<<<NB * (NPTS / 64),  128, 0, stream>>>(feat, ws);
  attn_kernel  <<<NB * (NPTS / 4),   128, 0, stream>>>(xyz, ws, attn);
  fc2_kernel   <<<NB * (NPTS / 64),  128, 0, stream>>>(feat, ws, out);
}
